// MinGRUTriton_77309411812
// MI455X (gfx1250) — compile-verified, matrix operand order checked by reading, not run on hardware
//
#include <hip/hip_runtime.h>
#include <hip/hip_bf16.h>
#include <math.h>

// MinGRU: B=4, L=8192, D=1024
//   g = sigmoid(x @ Wg^T + bg); c = tanh(x @ Wc^T + bc); h = scan(h = g*h + c)
//
// Pipeline:
//   K1 mingru_proj      : fused dual GEMM (WMMA bf16 hi/lo split, f32 acc),
//                         double-buffered LDS -> g in d_out, c in ws
//   K2 chunk scan       : per (b,d,chunk) local (prod g, local h)  -> Sa, Sh
//   K3 chunk prefix     : per (b,d) serial scan over 64 chunk summaries (in place)
//   K4 apply            : recompute chunk scan with correct h0 -> h overwrites d_out
//
// ws requirement: (B*L*D + 2*B*NC*D)*4 bytes ~= 136.3 MB.

#define B_DIM 4
#define L_DIM 8192
#define D_DIM 1024
#define NC    64
#define LC    (L_DIM / NC)
#define SK    40          // padded LDS row stride in bf16 elements (32 + 8)
#define STAGE 20480       // bf16 elements per LDS stage (40 KB)

typedef __attribute__((ext_vector_type(16))) __bf16 v16bf;
typedef __attribute__((ext_vector_type(8)))  float  v8f;

__device__ __forceinline__ v8f wmma_bf16(v16bf a, v16bf b, v8f c) {
  // (neg_a, A, neg_b, B, c_mod, C, reuse_a, reuse_b)
  return __builtin_amdgcn_wmma_f32_16x16x32_bf16(false, a, false, b, (short)0, c, false, false);
}

// Load one 16x32 bf16 A/B fragment from LDS per the CDNA5 16-bit operand layout:
// lanes 0-15 = row r, K in {0..7, 16..23}; lanes 16-31 = row r, K in {8..15, 24..31}.
// base already points at row*SK + (lane>=16 ? 8 : 0).
__device__ __forceinline__ v16bf ld_frag(const __bf16* __restrict__ p, int base) {
  union { uint4 u[2]; v16bf v; } t;
  t.u[0] = *(const uint4*)(p + base);        // K = kh*8 + 0..7
  t.u[1] = *(const uint4*)(p + base + 16);   // K = kh*8 + 16..23
  return t.v;
}

__device__ __forceinline__ void cvt_store4(__bf16* __restrict__ hi,
                                           __bf16* __restrict__ lo, float4 v) {
  float f[4] = {v.x, v.y, v.z, v.w};
#pragma unroll
  for (int i = 0; i < 4; ++i) {
    __bf16 h = (__bf16)f[i];
    hi[i] = h;
    lo[i] = (__bf16)(f[i] - (float)h);
  }
}

__device__ __forceinline__ float fast_sigmoid(float z) {
  return __builtin_amdgcn_rcpf(1.0f + __expf(-z));
}
__device__ __forceinline__ float fast_tanh(float z) {
  // 1 - 2/(1+e^{2z}); rcp(inf)=0 so both saturations are exact.
  return 1.0f - 2.0f * __builtin_amdgcn_rcpf(1.0f + __expf(2.0f * z));
}

// ---------------- Kernel 1: fused dual projection GEMM ----------------
// grid = (N/64, M/128), block = 256 (8 waves, 4x2). Block tile 128x64x32,
// double-buffered LDS (2 x 40 KB of 320 KB WGP LDS).
__global__ __launch_bounds__(256) void mingru_proj_kernel(
    const float* __restrict__ x, const float* __restrict__ Wg,
    const float* __restrict__ bg, const float* __restrict__ Wc,
    const float* __restrict__ bc, float* __restrict__ gate_out,
    float* __restrict__ cand_out) {
  __shared__ __bf16 smem[2][STAGE];

  const int tid  = threadIdx.x;
  const int lane = tid & 31;
  const int wid  = tid >> 5;
  const int wm   = wid >> 1;       // 0..3
  const int wn   = wid & 1;        // 0..1
  const int m0   = blockIdx.y * 128;
  const int n0   = blockIdx.x * 64;

  const int r  = lane & 15;
  const int kh = lane >> 4;
  const int ab0 = (wm * 32 + r) * SK + kh * 8;
  const int bb0 = (wn * 32 + r) * SK + kh * 8;

  v8f accg[2][2] = {};
  v8f accc[2][2] = {};

  // Stage one 128x32 x-tile + two 64x32 W-tiles (f32 -> bf16 hi/lo) into buffer s.
  auto stage = [&](int s, int kb) {
    __bf16* Xh = smem[s];
    __bf16* Xl = smem[s] + 5120;
    __bf16* Gh = smem[s] + 10240;
    __bf16* Gl = smem[s] + 12800;
    __bf16* Ch = smem[s] + 15360;
    __bf16* Cl = smem[s] + 17920;
#pragma unroll
    for (int i = 0; i < 4; ++i) {
      int q = tid + i * 256;             // 1024 float4 loads for x
      int row = q >> 3;
      int k4  = (q & 7) << 2;
      float4 v = *(const float4*)(x + (size_t)(m0 + row) * D_DIM + kb + k4);
      cvt_store4(Xh + row * SK + k4, Xl + row * SK + k4, v);
    }
#pragma unroll
    for (int i = 0; i < 2; ++i) {
      int q = tid + i * 256;             // 512 float4 loads per W matrix
      int row = q >> 3;
      int k4  = (q & 7) << 2;
      size_t widx = (size_t)(n0 + row) * D_DIM + kb + k4;
      float4 gv = *(const float4*)(Wg + widx);
      float4 cv = *(const float4*)(Wc + widx);
      cvt_store4(Gh + row * SK + k4, Gl + row * SK + k4, gv);
      cvt_store4(Ch + row * SK + k4, Cl + row * SK + k4, cv);
    }
  };

  stage(0, 0);
  int cur = 0;

  for (int kb = 0; kb < D_DIM; kb += 32) {
    __syncthreads();
    if (kb + 32 < D_DIM) stage(cur ^ 1, kb + 32);   // overlap next stage with WMMAs

    const __bf16* Xh = smem[cur];
    const __bf16* Xl = smem[cur] + 5120;
    const __bf16* Gh = smem[cur] + 10240;
    const __bf16* Gl = smem[cur] + 12800;
    const __bf16* Ch = smem[cur] + 15360;
    const __bf16* Cl = smem[cur] + 17920;

    v16bf ah[2], al[2];
#pragma unroll
    for (int mt = 0; mt < 2; ++mt) {
      ah[mt] = ld_frag(Xh, ab0 + mt * 16 * SK);
      al[mt] = ld_frag(Xl, ab0 + mt * 16 * SK);
    }
    v16bf bgh[2], bgl[2], bch[2], bcl[2];
#pragma unroll
    for (int nt = 0; nt < 2; ++nt) {
      bgh[nt] = ld_frag(Gh, bb0 + nt * 16 * SK);
      bgl[nt] = ld_frag(Gl, bb0 + nt * 16 * SK);
      bch[nt] = ld_frag(Ch, bb0 + nt * 16 * SK);
      bcl[nt] = ld_frag(Cl, bb0 + nt * 16 * SK);
    }
    // hi/lo split: A*B ~= Ah*Bh + Ah*Bl + Al*Bh  (recovers ~f32 precision)
#pragma unroll
    for (int mt = 0; mt < 2; ++mt)
#pragma unroll
      for (int nt = 0; nt < 2; ++nt) {
        accg[mt][nt] = wmma_bf16(ah[mt], bgh[nt], accg[mt][nt]);
        accg[mt][nt] = wmma_bf16(ah[mt], bgl[nt], accg[mt][nt]);
        accg[mt][nt] = wmma_bf16(al[mt], bgh[nt], accg[mt][nt]);
        accc[mt][nt] = wmma_bf16(ah[mt], bch[nt], accc[mt][nt]);
        accc[mt][nt] = wmma_bf16(ah[mt], bcl[nt], accc[mt][nt]);
        accc[mt][nt] = wmma_bf16(al[mt], bch[nt], accc[mt][nt]);
      }
    cur ^= 1;
  }

  // Epilogue: C/D layout -> v8f element rr holds row (rr + 8*kh), col = lane&15
#pragma unroll
  for (int mt = 0; mt < 2; ++mt)
#pragma unroll
    for (int nt = 0; nt < 2; ++nt) {
      int n = n0 + wn * 32 + nt * 16 + r;
      int mbase = m0 + wm * 32 + mt * 16 + kh * 8;
      float bgv = bg[n];
      float bcv = bc[n];
#pragma unroll
      for (int rr = 0; rr < 8; ++rr) {
        size_t oidx = (size_t)(mbase + rr) * D_DIM + n;
        gate_out[oidx] = fast_sigmoid(accg[mt][nt][rr] + bgv);
        cand_out[oidx] = fast_tanh(accc[mt][nt][rr] + bcv);
      }
    }
}

// ---------------- Kernel 2: per-chunk local scan summaries ----------------
// grid = (D/256, NC, B). Thread owns channel d of chunk: (prod g, local h).
__global__ __launch_bounds__(256) void mingru_chunk_scan_kernel(
    const float* __restrict__ g, const float* __restrict__ c,
    float* __restrict__ Sa, float* __restrict__ Sh) {
  const int d = blockIdx.x * 256 + threadIdx.x;
  const int chunk = blockIdx.y;
  const int b = blockIdx.z;
  float a = 1.0f, h = 0.0f;
  size_t base = ((size_t)b * L_DIM + (size_t)chunk * LC) * D_DIM + d;
#pragma unroll 4
  for (int l = 0; l < LC; ++l) {
    size_t idx = base + (size_t)l * D_DIM;
    float gv = __builtin_nontemporal_load(g + idx);   // streamed, >L2 reuse distance
    float cv = __builtin_nontemporal_load(c + idx);
    a *= gv;
    h = gv * h + cv;
  }
  size_t s = ((size_t)b * NC + chunk) * D_DIM + d;
  Sa[s] = a;
  Sh[s] = h;
}

// ---------------- Kernel 3: inter-chunk exclusive scan (in place) ----------------
// grid = (D/256, 1, B). Sh[chunk] := h at chunk start.
__global__ __launch_bounds__(256) void mingru_chunk_prefix_kernel(
    const float* __restrict__ Sa, float* __restrict__ Sh) {
  const int d = blockIdx.x * 256 + threadIdx.x;
  const int b = blockIdx.z;
  float h0 = 0.0f;
  for (int chunk = 0; chunk < NC; ++chunk) {
    size_t s = ((size_t)b * NC + chunk) * D_DIM + d;
    float a  = Sa[s];
    float hl = Sh[s];
    Sh[s] = h0;          // exclusive prefix = initial h for this chunk
    h0 = a * h0 + hl;
  }
}

// ---------------- Kernel 4: apply pass, overwrite gates with h ----------------
// grid = (D/256, NC, B). Reads g (d_out) & c (ws), writes h into d_out in place.
__global__ __launch_bounds__(256) void mingru_apply_kernel(
    float* __restrict__ gh, const float* __restrict__ c,
    const float* __restrict__ Sh) {
  const int d = blockIdx.x * 256 + threadIdx.x;
  const int chunk = blockIdx.y;
  const int b = blockIdx.z;
  float h = Sh[((size_t)b * NC + chunk) * D_DIM + d];
  size_t base = ((size_t)b * L_DIM + (size_t)chunk * LC) * D_DIM + d;
#pragma unroll 4
  for (int l = 0; l < LC; ++l) {
    size_t idx = base + (size_t)l * D_DIM;
    float gv = __builtin_nontemporal_load(gh + idx);
    float cv = __builtin_nontemporal_load(c + idx);
    h = gv * h + cv;
    __builtin_nontemporal_store(h, gh + idx);         // h is final output, streamed
  }
}

extern "C" void kernel_launch(void* const* d_in, const int* in_sizes, int n_in,
                              void* d_out, int out_size, void* d_ws, size_t ws_size,
                              hipStream_t stream) {
  const float* x  = (const float*)d_in[0];
  const float* Wg = (const float*)d_in[1];
  const float* bg = (const float*)d_in[2];
  const float* Wc = (const float*)d_in[3];
  const float* bc = (const float*)d_in[4];
  float* out = (float*)d_out;

  float* cand = (float*)d_ws;                                   // B*L*D floats
  float* Sa   = cand + (size_t)B_DIM * L_DIM * D_DIM;           // B*NC*D floats
  float* Sh   = Sa + (size_t)B_DIM * NC * D_DIM;                // B*NC*D floats

  dim3 g1(D_DIM / 64, (B_DIM * L_DIM) / 128, 1);                // (16, 256)
  mingru_proj_kernel<<<g1, 256, 0, stream>>>(x, Wg, bg, Wc, bc, out, cand);

  dim3 g2(D_DIM / 256, NC, B_DIM);                              // (4, 64, 4)
  mingru_chunk_scan_kernel<<<g2, 256, 0, stream>>>(out, cand, Sa, Sh);

  dim3 g3(D_DIM / 256, 1, B_DIM);                               // (4, 1, 4)
  mingru_chunk_prefix_kernel<<<g3, 256, 0, stream>>>(Sa, Sh);

  mingru_apply_kernel<<<g2, 256, 0, stream>>>(out, cand, Sh);
}
